// HGatedDeltaNet_45732811768014
// MI455X (gfx1250) — matrix operand order repaired and checked by reading, not compiled, not run on hardware
//
#include <hip/hip_runtime.h>
#include <hip/hip_bf16.h>
#include <math.h>

// ---------------------------------------------------------------------------
// HGatedDeltaNet forward for MI455X (gfx1250, wave32).
//   * bf16 WMMA (v_wmma_f32_16x16x32_bf16) for all dense projections
//   * register-resident Fenwick delta-rule scan: 768 independent wave32
//     chains (6 heads x 128 v-rows), 14x64 state = 28 VGPRs/lane
//   * cross-lane dots via __shfl_xor butterflies; global_prefetch_b8 for
//     next-step operands
//   * GEMM N-edge handled by clamping the OOB lane's weight row (loads stay
//     branch-free; only stores are guarded) so the K-loop is a pure
//     load-clause + v_wmma pipeline with no per-iteration EXEC save/restore.
// ---------------------------------------------------------------------------

#define TT  512
#define HID 768
#define HH  6
#define DKD 64
#define DVD 128
#define NLV 14
#define KDIM 384
#define VDIM 768

typedef __bf16 bf16_t;
typedef __attribute__((ext_vector_type(16))) __bf16 v16bf;
typedef __attribute__((ext_vector_type(8)))  __bf16 v8bf;
typedef __attribute__((ext_vector_type(8)))  float  v8f;

// ---------------------------------------------------------------- fp32->bf16
__global__ void hdn_f32_to_bf16(const float* __restrict__ in,
                                bf16_t* __restrict__ out, int n) {
  int i = blockIdx.x * blockDim.x + threadIdx.x;
  if (i < n) out[i] = (bf16_t)in[i];
}

// ------------------------------------------------------------ WMMA bf16 GEMM
// C[M,N] (f32, row stride N) = A[M,K] (bf16) * W[N,K]^T (bf16).
// One wave per 16x16 tile; K stepped by 32 with v_wmma_f32_16x16x32_bf16.
__global__ __launch_bounds__(32)
void hdn_wmma_gemm(const bf16_t* __restrict__ A, const bf16_t* __restrict__ W,
                   float* __restrict__ C, int M, int N, int K) {
  const int m0   = blockIdx.x * 16;
  const int n0   = blockIdx.y * 16;
  const int lane = threadIdx.x;
  const int hi   = lane >> 4;     // half-wave selector
  const int l15  = lane & 15;

  v8f acc = {};
  const int m  = m0 + l15;        // A row owned by this lane
  const int nB = n0 + l15;        // W row (= C column) owned by this lane
  const bool nOK = nB < N;
  const int nC = nOK ? nB : (N - 1);   // clamp: OOB lanes load a valid row;
                                       // they only feed unstored D columns.

  const bf16_t* Arow = A + (size_t)m * K + hi * 8;
  const bf16_t* Wrow = W + (size_t)nC * K + hi * 8;

  for (int k0 = 0; k0 < K; k0 += 32) {
    // A fragment: lane holds M=l15, K in {hi*8..hi*8+7} U {16+hi*8..23+hi*8}
    v8bf alo = *(const v8bf*)(Arow + k0);
    v8bf ahi = *(const v8bf*)(Arow + k0 + 16);
    // B fragment: lane holds N=l15, K in {hi*8..hi*8+7} U {16+hi*8..23+hi*8}
    v8bf blo = *(const v8bf*)(Wrow + k0);
    v8bf bhi = *(const v8bf*)(Wrow + k0 + 16);
    v16bf a, b;
#pragma unroll
    for (int e = 0; e < 8; ++e) {
      a[e] = alo[e]; a[e + 8] = ahi[e];
      b[e] = blo[e]; b[e + 8] = bhi[e];
    }
    acc = __builtin_amdgcn_wmma_f32_16x16x32_bf16(
        /*neg_a=*/false, a, /*neg_b=*/false, b,
        /*c_mod=*/(short)0, acc, /*reuse_a=*/false, /*reuse_b=*/false);
  }
  if (nOK) {
#pragma unroll
    for (int v = 0; v < 8; ++v) {
      int row = m0 + v + hi * 8;               // D layout: M = v + hi*8
      C[(size_t)row * N + nB] = acc[v];
    }
  }
}

// ----------------------------------------------- causal depthwise conv + SiLU
__global__ void hdn_conv_silu(const float* __restrict__ x,
                              const float* __restrict__ w,
                              float* __restrict__ y, int C) {
  int i = blockIdx.x * blockDim.x + threadIdx.x;
  if (i >= TT * C) return;
  int t = i / C, c = i - t * C;
  float acc = 0.f;
#pragma unroll
  for (int j = 0; j < 4; ++j) {
    int ts = t - 3 + j;
    if (ts >= 0) acc += w[c * 4 + j] * x[(size_t)ts * C + c];
  }
  y[i] = acc / (1.f + expf(-acc));  // silu
}

// --------------------------------------------- per-(t,h) L2 norm over DK=64
__global__ __launch_bounds__(32)
void hdn_l2norm(float* __restrict__ buf, float extraScale) {
  int row = blockIdx.x;             // t*H + h, rows = T*H
  int lane = threadIdx.x;
  float* p = buf + (size_t)row * DKD;
  float x0 = p[lane], x1 = p[lane + 32];
  float ss = x0 * x0 + x1 * x1;
#pragma unroll
  for (int off = 16; off > 0; off >>= 1) ss += __shfl_xor(ss, off, 32);
  float rs = rsqrtf(ss + 1e-6f) * extraScale;
  p[lane] = x0 * rs;
  p[lane + 32] = x1 * rs;
}

// -------------------------------------------------- alpha/beta gate scalars
__global__ void hdn_gates(const float* __restrict__ hs,
                          const float* __restrict__ Wa,
                          const float* __restrict__ Wb,
                          const float* __restrict__ A_log,
                          const float* __restrict__ dt_bias,
                          float* __restrict__ alpha,
                          float* __restrict__ beta) {
  int i = blockIdx.x * blockDim.x + threadIdx.x;
  if (i >= TT * HH) return;
  int t = i / HH, h = i - t * HH;
  const float* x  = hs + (size_t)t * HID;
  const float* wa = Wa + (size_t)h * HID;
  const float* wb = Wb + (size_t)h * HID;
  float da = 0.f, db = 0.f;
  for (int k = 0; k < HID; ++k) { da += x[k] * wa[k]; db += x[k] * wb[k]; }
  float z  = da + dt_bias[h];
  float sp = (z > 20.f) ? z : log1pf(expf(z));          // softplus
  alpha[i] = expf(-expf(A_log[h]) * sp);
  beta[i]  = 1.f / (1.f + expf(-db));                   // sigmoid
}

// ---------------------------------------------------- lambda level weights
__global__ void hdn_lambda(const float* __restrict__ lproj,
                           const float* __restrict__ Lp,
                           float* __restrict__ lam) {
  int i = blockIdx.x * blockDim.x + threadIdx.x;
  if (i >= TT * HH * NLV) return;
  int rem = i % (HH * NLV);
  float z = Lp[rem] * lproj[i];
  lam[i] = (z > 20.f) ? z : log1pf(expf(z));            // softplus
}

// ------------------------------------------------ hierarchical delta scan
// One wave32 per (head, v-row). Lane owns DK columns {lane, lane+32} of all
// NL levels -> S lives in 28 VGPRs/lane. Dots reduced with shfl_xor.
__global__ __launch_bounds__(32)
void hdn_scan(const float* __restrict__ q, const float* __restrict__ k,
              const float* __restrict__ v, const float* __restrict__ alpha,
              const float* __restrict__ beta, const float* __restrict__ lam,
              float* __restrict__ o) {
  const int wid  = blockIdx.x;       // 0..767
  const int h    = wid >> 7;
  const int vr   = wid & 127;
  const int lane = threadIdx.x;

  float S[NLV][2];
#pragma unroll
  for (int l = 0; l < NLV; ++l) { S[l][0] = 0.f; S[l][1] = 0.f; }

  for (int t = 0; t < TT; ++t) {
    const int rh = t * HH + h;
    const float* kp = k + (size_t)rh * DKD;
    const float* qp = q + (size_t)rh * DKD;
    const float kc0 = kp[lane], kc1 = kp[lane + 32];
    const float qc0 = qp[lane], qc1 = qp[lane + 32];
    const float vt  = v[(size_t)rh * DVD + vr];
    const float al  = alpha[rh];
    const float be  = beta[rh];
    if (t + 1 < TT) {                          // next-step operand prefetch
      __builtin_prefetch(kp + HH * DKD, 0, 1);
      __builtin_prefetch(qp + HH * DKD, 0, 1);
    }
    // Sk = S . k per level (old S), broadcast to all lanes
    float d[NLV];
#pragma unroll
    for (int l = 0; l < NLV; ++l) {
      float p = S[l][0] * kc0 + S[l][1] * kc1;
#pragma unroll
      for (int off = 16; off > 0; off >>= 1) p += __shfl_xor(p, off, 32);
      d[l] = p;
    }
    // gated delta transition: S = alpha*(S - beta*(S.k) k^T)
#pragma unroll
    for (int l = 0; l < NLV; ++l) {
      S[l][0] = al * (S[l][0] - be * d[l] * kc0);
      S[l][1] = al * (S[l][1] - be * d[l] * kc1);
    }
    // rank-1 token insert into level 0
    const float bv = be * vt;
    S[0][0] += bv * kc0;
    S[0][1] += bv * kc1;
    // Fenwick merge: levels < j collapse into level j (j = trailing ones of t)
    const int j = __builtin_ctz(~t);
    if (j > 0) {
      float m0 = 0.f, m1 = 0.f;
#pragma unroll
      for (int l = 0; l < NLV; ++l)
        if (l < j) { m0 += S[l][0]; m1 += S[l][1]; S[l][0] = 0.f; S[l][1] = 0.f; }
#pragma unroll
      for (int l = 1; l < NLV; ++l)
        if (l == j) { S[l][0] += m0; S[l][1] += m1; }
    }
    // readout: o = sum_l lam_l * (S^l q)
    const float* lp = lam + (size_t)rh * NLV;
    float r = 0.f;
#pragma unroll
    for (int l = 0; l < NLV; ++l) r += lp[l] * (S[l][0] * qc0 + S[l][1] * qc1);
#pragma unroll
    for (int off = 16; off > 0; off >>= 1) r += __shfl_xor(r, off, 32);
    if (lane == 0) o[(size_t)rh * DVD + vr] = r;
  }
}

// ---------------------------------- gated RMSNorm -> bf16 for output GEMM
__global__ __launch_bounds__(32)
void hdn_rms_gate(const float* __restrict__ o, const float* __restrict__ g,
                  const float* __restrict__ rms_w, bf16_t* __restrict__ out) {
  int row  = blockIdx.x;           // t*H + h
  int lane = threadIdx.x;
  const float* op = o + (size_t)row * DVD;
  const float* gp = g + (size_t)row * DVD;
  float x[4];
  float ss = 0.f;
#pragma unroll
  for (int jj = 0; jj < 4; ++jj) { x[jj] = op[lane + 32 * jj]; ss += x[jj] * x[jj]; }
#pragma unroll
  for (int off = 16; off > 0; off >>= 1) ss += __shfl_xor(ss, off, 32);
  float rs = rsqrtf(ss * (1.f / DVD) + 1e-5f);
#pragma unroll
  for (int jj = 0; jj < 4; ++jj) {
    int c = lane + 32 * jj;
    float gv  = gp[c];
    float sil = gv / (1.f + expf(-gv));
    out[(size_t)row * DVD + c] = (bf16_t)(x[jj] * rs * rms_w[c] * sil);
  }
}

// ---------------------------------------------------------------------------
extern "C" void kernel_launch(void* const* d_in, const int* in_sizes, int n_in,
                              void* d_out, int out_size, void* d_ws, size_t ws_size,
                              hipStream_t stream) {
  (void)in_sizes; (void)n_in; (void)out_size; (void)ws_size;
  const float* hs      = (const float*)d_in[0];
  const float* Wq      = (const float*)d_in[1];
  const float* Wk      = (const float*)d_in[2];
  const float* Wv      = (const float*)d_in[3];
  const float* Wa      = (const float*)d_in[4];
  const float* Wb      = (const float*)d_in[5];
  const float* Wg      = (const float*)d_in[6];
  const float* Wl      = (const float*)d_in[7];
  const float* Wo      = (const float*)d_in[8];
  const float* conv_q  = (const float*)d_in[9];
  const float* conv_k  = (const float*)d_in[10];
  const float* conv_v  = (const float*)d_in[11];
  const float* A_log   = (const float*)d_in[12];
  const float* dt_bias = (const float*)d_in[13];
  const float* L_param = (const float*)d_in[14];
  const float* rms_w   = (const float*)d_in[15];
  float* out = (float*)d_out;

  char* base = (char*)d_ws;
  size_t off = 0;
  auto alloc = [&](size_t nbytes) -> void* {
    off = (off + 255) & ~(size_t)255;
    void* p = base + off;
    off += nbytes;
    return p;
  };

  bf16_t* hs_bf = (bf16_t*)alloc((size_t)TT * HID * 2);
  bf16_t* Wq_bf = (bf16_t*)alloc((size_t)KDIM * HID * 2);
  bf16_t* Wk_bf = (bf16_t*)alloc((size_t)KDIM * HID * 2);
  bf16_t* Wv_bf = (bf16_t*)alloc((size_t)VDIM * HID * 2);
  bf16_t* Wg_bf = (bf16_t*)alloc((size_t)VDIM * HID * 2);
  bf16_t* Wl_bf = (bf16_t*)alloc((size_t)HH * NLV * HID * 2);
  bf16_t* Wo_bf = (bf16_t*)alloc((size_t)HID * VDIM * 2);
  float* qpre   = (float*)alloc((size_t)TT * KDIM * 4);
  float* kpre   = (float*)alloc((size_t)TT * KDIM * 4);
  float* vpre   = (float*)alloc((size_t)TT * VDIM * 4);
  float* qbuf   = (float*)alloc((size_t)TT * KDIM * 4);
  float* kbuf   = (float*)alloc((size_t)TT * KDIM * 4);
  float* vbuf   = (float*)alloc((size_t)TT * VDIM * 4);
  float* gbuf   = (float*)alloc((size_t)TT * VDIM * 4);
  float* lproj  = (float*)alloc((size_t)TT * HH * NLV * 4);
  float* alphab = (float*)alloc((size_t)TT * HH * 4);
  float* betab  = (float*)alloc((size_t)TT * HH * 4);
  float* lamb   = (float*)alloc((size_t)TT * HH * NLV * 4);
  float* obuf   = (float*)alloc((size_t)TT * VDIM * 4);
  bf16_t* og_bf = (bf16_t*)alloc((size_t)TT * VDIM * 2);

  auto cvt = [&](const float* src, bf16_t* dst, int cnt) {
    hdn_f32_to_bf16<<<dim3((cnt + 255) / 256), dim3(256), 0, stream>>>(src, dst, cnt);
  };
  cvt(hs, hs_bf, TT * HID);
  cvt(Wq, Wq_bf, KDIM * HID);
  cvt(Wk, Wk_bf, KDIM * HID);
  cvt(Wv, Wv_bf, VDIM * HID);
  cvt(Wg, Wg_bf, VDIM * HID);
  cvt(Wl, Wl_bf, HH * NLV * HID);
  cvt(Wo, Wo_bf, HID * VDIM);

  auto gemm = [&](const bf16_t* A, const bf16_t* W, float* C, int M, int N, int K) {
    hdn_wmma_gemm<<<dim3(M / 16, (N + 15) / 16), dim3(32), 0, stream>>>(A, W, C, M, N, K);
  };
  gemm(hs_bf, Wq_bf, qpre, TT, KDIM, HID);
  gemm(hs_bf, Wk_bf, kpre, TT, KDIM, HID);
  gemm(hs_bf, Wv_bf, vpre, TT, VDIM, HID);
  gemm(hs_bf, Wg_bf, gbuf, TT, VDIM, HID);
  gemm(hs_bf, Wl_bf, lproj, TT, HH * NLV, HID);

  hdn_conv_silu<<<dim3((TT * KDIM + 255) / 256), dim3(256), 0, stream>>>(qpre, conv_q, qbuf, KDIM);
  hdn_conv_silu<<<dim3((TT * KDIM + 255) / 256), dim3(256), 0, stream>>>(kpre, conv_k, kbuf, KDIM);
  hdn_conv_silu<<<dim3((TT * VDIM + 255) / 256), dim3(256), 0, stream>>>(vpre, conv_v, vbuf, VDIM);

  hdn_l2norm<<<dim3(TT * HH), dim3(32), 0, stream>>>(qbuf, 0.125f);  // DK^-0.5
  hdn_l2norm<<<dim3(TT * HH), dim3(32), 0, stream>>>(kbuf, 1.0f);

  hdn_gates<<<dim3((TT * HH + 127) / 128), dim3(128), 0, stream>>>(
      hs, Wa, Wb, A_log, dt_bias, alphab, betab);
  hdn_lambda<<<dim3((TT * HH * NLV + 255) / 256), dim3(256), 0, stream>>>(
      lproj, L_param, lamb);

  hdn_scan<<<dim3(HH * DVD), dim3(32), 0, stream>>>(
      qbuf, kbuf, vbuf, alphab, betab, lamb, obuf);

  hdn_rms_gate<<<dim3(TT * HH), dim3(32), 0, stream>>>(obuf, gbuf, rms_w, og_bf);

  gemm(og_bf, Wo_bf, out, TT, HID, VDIM);
}
